// DeepseekVL2MoEGateAdapter_44418551775974
// MI455X (gfx1250) — compile-verified, matrix operand order checked by reading, not run on hardware
//
#include <hip/hip_runtime.h>
#include <hip/hip_bf16.h>
#include <math.h>

#define H     7168
#define NEXP  256
#define KC    32
#define KT    (H / KC)      // 224 k-chunks
#define BT    64            // tokens per workgroup
#define APAD  40            // padded bf16 row stride for staged A (80B: spreads LDS banks)
#define SPAD  257           // padded f32 row stride for score matrix
#define TOPK  8
#define NTILES (NEXP / 16)  // 16 expert tiles

typedef __attribute__((ext_vector_type(4)))  float  v4f;
typedef __attribute__((ext_vector_type(8)))  float  v8f;
typedef __attribute__((ext_vector_type(8)))  __bf16 v8bf;
typedef __attribute__((ext_vector_type(16))) __bf16 v16bf;

// split 8 fp32 -> bf16 hi + bf16 lo (lo = residual, recovers ~16 mantissa bits)
static __device__ __forceinline__ void cvt_hilo8(v4f f0, v4f f1, v8bf& hi, v8bf& lo) {
#pragma unroll
  for (int i = 0; i < 4; ++i) {
    __bf16 h0 = (__bf16)f0[i];
    hi[i] = h0;
    lo[i] = (__bf16)(f0[i] - (float)h0);
    __bf16 h1 = (__bf16)f1[i];
    hi[i + 4] = h1;
    lo[i + 4] = (__bf16)(f1[i] - (float)h1);
  }
}

static __device__ __forceinline__ void cvt_hilo16(v4f r0, v4f r1, v4f r2, v4f r3,
                                                  v16bf& hi, v16bf& lo) {
#pragma unroll
  for (int i = 0; i < 4; ++i) {
    __bf16 h;
    h = (__bf16)r0[i]; hi[i]      = h; lo[i]      = (__bf16)(r0[i] - (float)h);
    h = (__bf16)r1[i]; hi[i + 4]  = h; lo[i + 4]  = (__bf16)(r1[i] - (float)h);
    h = (__bf16)r2[i]; hi[i + 8]  = h; lo[i + 8]  = (__bf16)(r2[i] - (float)h);
    h = (__bf16)r3[i]; hi[i + 12] = h; lo[i + 12] = (__bf16)(r3[i] - (float)h);
  }
}

// ---------------------------------------------------------------------------
// Pre-convert gating weight into WMMA-fragment-ordered bf16 hi/lo:
// whi/wlo layout: [tile(16)][kt(224)][lane(32)][16 bf16]  -> per-lane 32B runs
// ---------------------------------------------------------------------------
__global__ __launch_bounds__(256)
void moe_wconv_kernel(const float* __restrict__ wgt,
                      __bf16* __restrict__ whi,
                      __bf16* __restrict__ wlo) {
  const int tid  = threadIdx.x;
  const int lane = tid & 31;
  const int wv   = tid >> 5;
  const int chunk = blockIdx.x * 8 + wv;        // 0 .. NTILES*KT-1
  const int ntab  = chunk / KT;                 // expert tile
  const int kt    = chunk % KT;                 // k-chunk
  const int col   = lane & 15;
  const int hh    = lane >> 4;

  const float* wp = wgt + (size_t)(ntab * 16 + col) * H + kt * KC + hh * 8;
  v4f r0 = *(const v4f*)(wp + 0);
  v4f r1 = *(const v4f*)(wp + 4);
  v4f r2 = *(const v4f*)(wp + 16);
  v4f r3 = *(const v4f*)(wp + 20);
  v16bf hi, lo;
  cvt_hilo16(r0, r1, r2, r3, hi, lo);

  const size_t off = ((size_t)chunk * 32 + lane) * 16;
  *(v16bf*)&whi[off] = hi;
  *(v16bf*)&wlo[off] = lo;
}

// ---------------------------------------------------------------------------
// Fused gating GEMM (bf16-split WMMA) + sigmoid + grouped top-k routing
// ---------------------------------------------------------------------------
template <bool PRECONV>
__global__ __launch_bounds__(256)
void moe_gate_kernel(const float* __restrict__ x,
                     const float* __restrict__ wgt,
                     const float* __restrict__ bias,
                     const __bf16* __restrict__ whi,
                     const __bf16* __restrict__ wlo,
                     int*   __restrict__ out_idx,
                     float* __restrict__ out_w,
                     int T) {
  __shared__ alignas(16) __bf16 sAhi[2][BT * APAD];
  __shared__ alignas(16) __bf16 sAlo[2][BT * APAD];
  __shared__ alignas(16) float  sScore[BT * SPAD];
  __shared__ float sBias[NEXP];

  const int tid  = threadIdx.x;
  const int lane = tid & 31;
  const int wv   = tid >> 5;       // wave id 0..7 -> expert slice
  const int col  = lane & 15;      // M (A rows) / N (B cols) within a 16-wide tile
  const int hh   = lane >> 4;      // lane-half selects K sub-pattern
  const int t0   = blockIdx.x * BT;
  const int ebase = wv * 32;

  // ---------------- prologue: bias + stage k-chunk 0 ----------------
  sBias[tid] = bias[tid];
  const int tokS = tid >> 2;       // 64 tokens
  const int qS   = tid & 3;        // 4 chunks of 8 floats = KC
  {
    const float* p = x + (size_t)(t0 + tokS) * H + qS * 8;
    v4f f0 = *(const v4f*)p;
    v4f f1 = *(const v4f*)(p + 4);
    v8bf hi, lo;
    cvt_hilo8(f0, f1, hi, lo);
    *(v8bf*)&sAhi[0][tokS * APAD + qS * 8] = hi;
    *(v8bf*)&sAlo[0][tokS * APAD + qS * 8] = lo;
  }
  __syncthreads();

  v8f acc[4][2] = {};

  // ---------------- main K loop ----------------
  for (int kt = 0; kt < KT; ++kt) {
    const int cur = kt & 1;
    const int kc  = kt * KC;

    // B fragments for this k-chunk
    v16bf bhi[2], blo[2];
#pragma unroll
    for (int nt = 0; nt < 2; ++nt) {
      if (PRECONV) {
        // fragment-ordered bf16: one 32B contiguous run per lane, coalesced
        const size_t boff = (((size_t)(wv * 2 + nt) * KT + kt) * 32 + lane) * 16;
        bhi[nt] = *(const v16bf*)&whi[boff];
        blo[nt] = *(const v16bf*)&wlo[boff];
        if (kt + 1 < KT) {
          __builtin_prefetch(&whi[boff + 32 * 16], 0, 3);
          __builtin_prefetch(&wlo[boff + 32 * 16], 0, 3);
        }
      } else {
        const float* wp = wgt + (size_t)(ebase + nt * 16 + col) * H + kc + hh * 8;
        v4f r0 = *(const v4f*)(wp + 0);
        v4f r1 = *(const v4f*)(wp + 4);
        v4f r2 = *(const v4f*)(wp + 16);
        v4f r3 = *(const v4f*)(wp + 20);
        if (kt + 1 < KT) __builtin_prefetch(wp + KC, 0, 3);
        cvt_hilo16(r0, r1, r2, r3, bhi[nt], blo[nt]);
      }
    }

    // issue next A tile global loads early (latency hiding)
    v4f a0, a1;
    if (kt + 1 < KT) {
      const float* p = x + (size_t)(t0 + tokS) * H + (kc + KC) + qS * 8;
      a0 = *(const v4f*)p;
      a1 = *(const v4f*)(p + 4);
    }

    // compute: 4 m-tiles x 2 n-tiles x 3 WMMAs (hi*hi + hi*lo + lo*hi)
#pragma unroll
    for (int mt = 0; mt < 4; ++mt) {
      const int ab = (mt * 16 + col) * APAD + hh * 8;
      v8bf h0 = *(const v8bf*)&sAhi[cur][ab];
      v8bf h1 = *(const v8bf*)&sAhi[cur][ab + 16];
      v8bf l0 = *(const v8bf*)&sAlo[cur][ab];
      v8bf l1 = *(const v8bf*)&sAlo[cur][ab + 16];
      v16bf ahi = __builtin_shufflevector(h0, h1, 0,1,2,3,4,5,6,7,8,9,10,11,12,13,14,15);
      v16bf alo = __builtin_shufflevector(l0, l1, 0,1,2,3,4,5,6,7,8,9,10,11,12,13,14,15);
#pragma unroll
      for (int nt = 0; nt < 2; ++nt) {
        acc[mt][nt] = __builtin_amdgcn_wmma_f32_16x16x32_bf16(
            false, ahi, false, bhi[nt], (short)0, acc[mt][nt], false, false);
        acc[mt][nt] = __builtin_amdgcn_wmma_f32_16x16x32_bf16(
            false, ahi, false, blo[nt], (short)0, acc[mt][nt], false, false);
        acc[mt][nt] = __builtin_amdgcn_wmma_f32_16x16x32_bf16(
            false, alo, false, bhi[nt], (short)0, acc[mt][nt], false, false);
      }
    }

    // stage next k-chunk into the other buffer
    if (kt + 1 < KT) {
      v8bf hi, lo;
      cvt_hilo8(a0, a1, hi, lo);
      *(v8bf*)&sAhi[cur ^ 1][tokS * APAD + qS * 8] = hi;
      *(v8bf*)&sAlo[cur ^ 1][tokS * APAD + qS * 8] = lo;
    }
    __syncthreads();
  }

  // ---------------- spill logits to LDS score matrix ----------------
  // C/D layout: VGPR r -> token = mt*16 + 8*hh + r; N = col
#pragma unroll
  for (int mt = 0; mt < 4; ++mt)
#pragma unroll
    for (int nt = 0; nt < 2; ++nt) {
      const int e  = ebase + nt * 16 + col;
      const int tb = mt * 16 + hh * 8;
#pragma unroll
      for (int r = 0; r < 8; ++r)
        sScore[(tb + r) * SPAD + e] = acc[mt][nt][r];
    }
  __syncthreads();

  // ---------------- fused routing epilogue: one thread per token ----------------
  if (tid < BT) {
    float* row = &sScore[tid * SPAD];

    // pass 1: sigmoid (stored back), per-group top-2 sums
    float gs[8];
#pragma unroll
    for (int g = 0; g < 8; ++g) {
      float t1 = -3.0e38f, t2 = -3.0e38f;
      for (int j = 0; j < 32; ++j) {
        const int e = g * 32 + j;
        float s = 1.0f / (1.0f + __expf(-row[e]));
        row[e] = s;
        float v = s + sBias[e];
        if (v > t1)      { t2 = t1; t1 = v; }
        else if (v > t2) { t2 = v; }
      }
      gs[g] = t1 + t2;
    }

    // top-4 groups (strict > => lower index wins ties, matches lax.top_k)
    int gmask = 0;
#pragma unroll
    for (int r = 0; r < 4; ++r) {
      float best = -3.0e38f; int bg = 0;
#pragma unroll
      for (int g = 0; g < 8; ++g) {
        bool c = (((gmask >> g) & 1) == 0) && (gs[g] > best);
        best = c ? gs[g] : best;
        bg   = c ? g : bg;
      }
      gmask |= (1 << bg);
    }

    // top-8 experts over masked scores (masked -> 0.0, exactly as reference)
    float bv[8]; int bix[8];
#pragma unroll
    for (int j = 0; j < 8; ++j) { bv[j] = -3.0e38f; bix[j] = 0; }
    for (int e = 0; e < NEXP; ++e) {
      float v = ((gmask >> (e >> 5)) & 1) ? (row[e] + sBias[e]) : 0.0f;
      bool gt[8];
#pragma unroll
      for (int j = 0; j < 8; ++j) gt[j] = v > bv[j];
#pragma unroll
      for (int j = 7; j > 0; --j) {
        bv[j]  = gt[j] ? (gt[j - 1] ? bv[j - 1] : v) : bv[j];
        bix[j] = gt[j] ? (gt[j - 1] ? bix[j - 1] : e) : bix[j];
      }
      bv[0]  = gt[0] ? v : bv[0];
      bix[0] = gt[0] ? e : bix[0];
    }

    // gather sigmoid weights (no bias), normalize, scale by 2.5
    float wsum = 1e-20f;
    float wk[8];
#pragma unroll
    for (int j = 0; j < 8; ++j) { wk[j] = row[bix[j]]; wsum += wk[j]; }
    const float scale = 2.5f / wsum;

    const int tg = t0 + tid;
#pragma unroll
    for (int j = 0; j < 8; ++j) {
      out_idx[tg * TOPK + j] = bix[j];
      out_w  [tg * TOPK + j] = wk[j] * scale;
    }
  }
}

extern "C" void kernel_launch(void* const* d_in, const int* in_sizes, int n_in,
                              void* d_out, int out_size, void* d_ws, size_t ws_size,
                              hipStream_t stream) {
  const float* x    = (const float*)d_in[0];   // [4,4096,7168] f32
  const float* wgt  = (const float*)d_in[1];   // [256,7168]    f32
  const float* bias = (const float*)d_in[2];   // [256]         f32
  const int T = in_sizes[0] / H;               // 16384 tokens

  int*   out_idx = (int*)d_out;                      // first T*8: int32 indices
  float* out_w   = (float*)d_out + (size_t)T * TOPK; // next  T*8: f32 weights

  // bf16 hi/lo weight images in scratch, fragment-ordered
  const size_t wimg_elems = (size_t)NTILES * KT * 32 * 16;    // 1,835,008 bf16
  const size_t need = 2 * wimg_elems * sizeof(__bf16);        // 7.34 MB

  dim3 grid(T / BT), block(256);
  if (ws_size >= need) {
    __bf16* whi = (__bf16*)d_ws;
    __bf16* wlo = whi + wimg_elems;
    hipLaunchKernelGGL(moe_wconv_kernel, dim3(NTILES * KT / 8), block, 0, stream,
                       wgt, whi, wlo);
    hipLaunchKernelGGL((moe_gate_kernel<true>), grid, block, 0, stream,
                       x, wgt, bias, whi, wlo, out_idx, out_w, T);
  } else {
    hipLaunchKernelGGL((moe_gate_kernel<false>), grid, block, 0, stream,
                       x, wgt, bias, (const __bf16*)nullptr, (const __bf16*)nullptr,
                       out_idx, out_w, T);
  }
}
